// ScaledDotAttention_44306882625978
// MI455X (gfx1250) — compile-verified, matrix operand order checked by reading, not run on hardware
//
#include <hip/hip_runtime.h>
#include <hip/hip_bf16.h>

typedef __attribute__((ext_vector_type(16))) _Float16 v16h;
typedef __attribute__((ext_vector_type(8)))  float    v8f;

#define HW   4096
#define CDIM 64
#define ATT_SCALE 0.125f   // 64^-0.5

// ---------------------------------------------------------------------------
// CDNA5 async global->LDS copy, 16 bytes per lane, tracked by ASYNCcnt.
// lds_byte: wave-relative LDS byte address (low 32 bits of flat LDS pointer).
// ---------------------------------------------------------------------------
__device__ __forceinline__ void async_copy16(unsigned lds_byte, const void* gptr) {
    asm volatile("global_load_async_to_lds_b128 %0, %1, off"
                 :: "v"(lds_byte), "v"(gptr)
                 : "memory");
}

// xor-lane swap via single ds_swizzle_b32 (group-of-32: and=0x1f, xor=mask)
#define XOR_SWIZ(v, mask) \
    __int_as_float(__builtin_amdgcn_ds_swizzle(__float_as_int(v), (0x1f | ((mask) << 10))))

// ---------------------------------------------------------------------------
// Kernel 1: per-(b,c) InstanceNorm statistics over 4096 spatial elements
// ---------------------------------------------------------------------------
__global__ __launch_bounds__(256) void in_stats_kernel(const float* __restrict__ x,
                                                       float* __restrict__ mean,
                                                       float* __restrict__ rstd) {
    __shared__ float ssum[256];
    __shared__ float ssq[256];
    const int bc = blockIdx.x;                  // 0..255 = b*64 + c
    const float* xp = x + (size_t)bc * HW;
    float s = 0.f, sq = 0.f;
    for (int i = threadIdx.x; i < HW; i += 256) {
        float v = xp[i];
        s += v; sq += v * v;
    }
    ssum[threadIdx.x] = s; ssq[threadIdx.x] = sq;
    __syncthreads();
    for (int off = 128; off > 0; off >>= 1) {
        if (threadIdx.x < off) {
            ssum[threadIdx.x] += ssum[threadIdx.x + off];
            ssq[threadIdx.x]  += ssq[threadIdx.x + off];
        }
        __syncthreads();
    }
    if (threadIdx.x == 0) {
        float mu  = ssum[0] * (1.f / HW);
        float var = ssq[0] * (1.f / HW) - mu * mu;
        mean[bc] = mu;
        rstd[bc] = rsqrtf(var + 1e-5f);
    }
}

// ---------------------------------------------------------------------------
// Kernel 2: normalize + 1x1 conv QKV; q,k as f16 [B, HW, C]; v as f16 [B, C, HW]
// (V written pre-transposed so attention can async-DMA contiguous rows.)
// ---------------------------------------------------------------------------
__global__ __launch_bounds__(192) void qkv_kernel(const float* __restrict__ x,
                                                  const float* __restrict__ w_qkv,
                                                  const float* __restrict__ b_qkv,
                                                  const float* __restrict__ mean,
                                                  const float* __restrict__ rstd,
                                                  _Float16* __restrict__ q,
                                                  _Float16* __restrict__ k,
                                                  _Float16* __restrict__ vt) {
    __shared__ float h[CDIM];
    const int b = blockIdx.x >> 12;             // HW = 4096 blocks per batch
    const int n = blockIdx.x & (HW - 1);
    const int tid = threadIdx.x;
    if (tid < CDIM) {
        const int c = tid;
        float xv = x[((size_t)(b * CDIM + c)) * HW + n];
        h[c] = (xv - mean[b * CDIM + c]) * rstd[b * CDIM + c];
    }
    __syncthreads();
    const int o = tid;                          // 0..191
    float acc = b_qkv[o];
    #pragma unroll 8
    for (int c = 0; c < CDIM; ++c) acc += w_qkv[o * CDIM + c] * h[c];
    const _Float16 hv = (_Float16)acc;
    if (o < 64) {
        q[((size_t)(b * HW + n)) * CDIM + o] = hv;
    } else if (o < 128) {
        k[((size_t)(b * HW + n)) * CDIM + (o - 64)] = hv;
    } else {
        vt[((size_t)(b * CDIM + (o - 128))) * HW + n] = hv;   // transposed
    }
}

// ---------------------------------------------------------------------------
// Kernel 3: flash attention, WMMA f32_16x16x32_f16, double-buffered async
// global->LDS staging of K and V^T blocks.
// Grid: B * (HW/64) blocks, 128 threads (4 waves), each wave owns 16 q-rows.
// ---------------------------------------------------------------------------
__global__ __launch_bounds__(128) void attn_kernel(const _Float16* __restrict__ q,
                                                   const _Float16* __restrict__ k,
                                                   const _Float16* __restrict__ vt,
                                                   float* __restrict__ o) {
    // [buf][key][ch] and [buf][ch][key], padded row stride 80 halfs (160 B)
    __shared__ __align__(16) _Float16 k_lds[2][64 * 80];
    __shared__ __align__(16) _Float16 vt_lds[2][64 * 80];
    __shared__ __align__(16) _Float16 p_lds[4 * 16 * 64];

    const int b     = blockIdx.x >> 6;          // 64 query tiles per batch
    const int qtile = blockIdx.x & 63;
    const int tid   = threadIdx.x;
    const int wave  = tid >> 5;
    const int lane  = tid & 31;
    const int lrow  = lane & 15;
    const int lhalf = lane >> 4;

    const unsigned kbase[2]  = { (unsigned)(uintptr_t)&k_lds[0][0],
                                 (unsigned)(uintptr_t)&k_lds[1][0] };
    const unsigned vtbase[2] = { (unsigned)(uintptr_t)&vt_lds[0][0],
                                 (unsigned)(uintptr_t)&vt_lds[1][0] };

    // Q A-fragments (16 rows x 64 ch) held in registers for the whole kernel.
    const int qrow = qtile * 64 + wave * 16 + lrow;
    const _Float16* qbase = q + ((size_t)(b * HW + qrow)) * CDIM;
    const v16h a0 = *(const v16h*)(qbase + lhalf * 16);        // K 0..31
    const v16h a1 = *(const v16h*)(qbase + 32 + lhalf * 16);   // K 32..63

    v8f acc[4] = {};                           // output: 4 tiles of 16 channels
    float m[8], l[8];
    #pragma unroll
    for (int j = 0; j < 8; ++j) { m[j] = -1e30f; l[j] = 0.f; }

    _Float16* pw = p_lds + wave * 16 * 64;

    const char* kg0  = (const char*)(k  + ((size_t)b * HW) * CDIM);
    const char* vtg0 = (const char*)(vt + ((size_t)b * CDIM) * HW);

    // ---- issue async staging of one 64-key block (8 b128s per wave) ----
    auto stage_kv = [&](int kb, int buf) {
        const char* kg = kg0 + (size_t)kb * 64 * CDIM * 2;        // 8 KB block
        // K block: 64 rows x 128 B, 512 chunks of 16 B
        for (int i = tid; i < 512; i += 128) {
            const int key = i >> 3, c16 = i & 7;
            async_copy16(kbase[buf] + key * 160 + c16 * 16, kg + i * 16);
        }
        // V^T block: 64 ch-rows x 128 B (row stride HW*2 bytes in global)
        const char* vg = vtg0 + (size_t)kb * 64 * 2;
        for (int i = tid; i < 512; i += 128) {
            const int ch = i >> 3, k16 = i & 7;
            async_copy16(vtbase[buf] + ch * 160 + k16 * 16,
                         vg + (size_t)ch * (HW * 2) + k16 * 16);
        }
    };

    stage_kv(0, 0);

    for (int kb = 0; kb < HW / 64; ++kb) {
        const int cur = kb & 1;
        if (kb + 1 < HW / 64) {
            stage_kv(kb + 1, cur ^ 1);          // prefetch next block
            asm volatile("s_wait_asynccnt 8" ::: "memory");   // current landed
        } else {
            asm volatile("s_wait_asynccnt 0" ::: "memory");
        }
        __syncthreads();                        // all waves' staging visible

        const _Float16* kl = &k_lds[cur][0];
        const _Float16* vl = &vt_lds[cur][0];

        // ---- S = Q * K^T : 4 key sub-tiles of 16, K-dim 64 = 2 WMMA steps ----
        v8f s[4];
        #pragma unroll
        for (int t = 0; t < 4; ++t) {
            const int key = t * 16 + lrow;      // B-matrix column = key index
            v8f c = {};
            v16h b0 = *(const v16h*)(&kl[key * 80 + lhalf * 16]);
            c = __builtin_amdgcn_wmma_f32_16x16x32_f16(false, a0, false, b0,
                                                       (short)0, c, false, false);
            v16h b1 = *(const v16h*)(&kl[key * 80 + 32 + lhalf * 16]);
            c = __builtin_amdgcn_wmma_f32_16x16x32_f16(false, a1, false, b1,
                                                       (short)0, c, false, false);
            s[t] = c;
        }

        // ---- online softmax (row = VGPR index j + 8*lane_half) ----
        #pragma unroll
        for (int j = 0; j < 8; ++j) {
            float rmax = fmaxf(fmaxf(s[0][j], s[1][j]), fmaxf(s[2][j], s[3][j])) * ATT_SCALE;
            rmax = fmaxf(rmax, XOR_SWIZ(rmax, 1));
            rmax = fmaxf(rmax, XOR_SWIZ(rmax, 2));
            rmax = fmaxf(rmax, XOR_SWIZ(rmax, 4));
            rmax = fmaxf(rmax, XOR_SWIZ(rmax, 8));
            float mnew  = fmaxf(m[j], rmax);
            float alpha = __expf(m[j] - mnew);
            m[j] = mnew;
            l[j] *= alpha;
            #pragma unroll
            for (int t = 0; t < 4; ++t) acc[t][j] *= alpha;

            const int row = j + lhalf * 8;
            float rsum = 0.f;
            #pragma unroll
            for (int t = 0; t < 4; ++t) {
                float p = __expf(s[t][j] * ATT_SCALE - mnew);
                rsum += p;
                pw[row * 64 + t * 16 + lrow] = (_Float16)p;
            }
            rsum += XOR_SWIZ(rsum, 1);
            rsum += XOR_SWIZ(rsum, 2);
            rsum += XOR_SWIZ(rsum, 4);
            rsum += XOR_SWIZ(rsum, 8);
            l[j] += rsum;
        }

        // ---- O += P * V : A = P [16 x 64] from LDS, B = V^T staged in LDS ----
        #pragma unroll
        for (int kc = 0; kc < 2; ++kc) {
            v16h ap = *(const v16h*)(&pw[lrow * 64 + kc * 32 + lhalf * 16]);
            #pragma unroll
            for (int t = 0; t < 4; ++t) {
                const int ch = t * 16 + lrow;   // B-matrix column = out channel
                v16h bv = *(const v16h*)(&vl[ch * 80 + kc * 32 + lhalf * 16]);
                acc[t] = __builtin_amdgcn_wmma_f32_16x16x32_f16(false, ap, false, bv,
                                                                (short)0, acc[t], false, false);
            }
        }
        __syncthreads();   // done reading buf[cur]; next iter may overwrite it
    }

    // ---- normalize by row sums and write O [B, HW, C] f32 ----
    #pragma unroll
    for (int j = 0; j < 8; ++j) {
        const int row  = j + lhalf * 8;
        const int grow = qtile * 64 + wave * 16 + row;
        const float inv = 1.f / l[j];
        #pragma unroll
        for (int t = 0; t < 4; ++t) {
            o[((size_t)(b * HW + grow)) * CDIM + t * 16 + lrow] = acc[t][j] * inv;
        }
    }
}

// ---------------------------------------------------------------------------
// Kernel 4: 1x1 conv projection + residual, out [B, C, H, W] f32
// ---------------------------------------------------------------------------
__global__ __launch_bounds__(64) void proj_kernel(const float* __restrict__ x,
                                                  const float* __restrict__ w_proj,
                                                  const float* __restrict__ b_proj,
                                                  const float* __restrict__ o,
                                                  float* __restrict__ out) {
    __shared__ float h[CDIM];
    const int b = blockIdx.x >> 12;
    const int n = blockIdx.x & (HW - 1);
    const int c = threadIdx.x;
    h[c] = o[((size_t)(b * HW + n)) * CDIM + c];
    __syncthreads();
    float acc = b_proj[c];
    #pragma unroll 8
    for (int cc = 0; cc < CDIM; ++cc) acc += w_proj[c * CDIM + cc] * h[cc];
    const size_t idx = ((size_t)(b * CDIM + c)) * HW + n;
    out[idx] = x[idx] + acc;
}

// ---------------------------------------------------------------------------
extern "C" void kernel_launch(void* const* d_in, const int* in_sizes, int n_in,
                              void* d_out, int out_size, void* d_ws, size_t ws_size,
                              hipStream_t stream) {
    const float* x      = (const float*)d_in[0];
    const float* w_qkv  = (const float*)d_in[1];
    const float* b_qkv  = (const float*)d_in[2];
    const float* w_proj = (const float*)d_in[3];
    const float* b_proj = (const float*)d_in[4];
    float* out = (float*)d_out;

    const size_t B = 4, elems = B * HW * CDIM;   // 1,048,576 per tensor
    char* ws = (char*)d_ws;
    float*    mean = (float*)ws;                       // 256 f32
    float*    rstd = mean + 256;                       // 256 f32
    float*    obuf = (float*)(ws + 4096);              // 4 MB f32
    _Float16* q    = (_Float16*)(ws + 4096 + elems * sizeof(float));
    _Float16* k    = q + elems;                        // 2 MB each (f16)
    _Float16* vt   = k + elems;

    in_stats_kernel<<<B * CDIM, 256, 0, stream>>>(x, mean, rstd);
    qkv_kernel<<<B * HW, 192, 0, stream>>>(x, w_qkv, b_qkv, mean, rstd, q, k, vt);
    attn_kernel<<<B * (HW / 64), 128, 0, stream>>>(q, k, vt, obuf);
    proj_kernel<<<B * HW, 64, 0, stream>>>(x, w_proj, b_proj, obuf, out);
}
